// SWA_22299470201349
// MI455X (gfx1250) — matrix operand order repaired and checked by reading, not compiled, not run on hardware
//
#include <hip/hip_runtime.h>

// ---------------------------------------------------------------------------
// SWA attention layer for gfx1250 (MI455X): bf16 WMMA everywhere, f32 accum.
// Shapes: T=2048, D=2048, H=16, d=128, WS=2048, NKV=4096.
// GEMM B-tiles staged to LDS via async global->LDS DMA (double buffered).
// ---------------------------------------------------------------------------

typedef unsigned short u16;
typedef __attribute__((ext_vector_type(16))) __bf16 bf16x16;
typedef __attribute__((ext_vector_type(8)))  __bf16 bf16x8;
typedef __attribute__((ext_vector_type(8)))  float  f32x8;
typedef int i32x4v __attribute__((vector_size(16)));   // <4 x i32>, matches builtin proto

#define DEV static __device__ __forceinline__

#if __has_builtin(__builtin_amdgcn_global_load_async_to_lds_b128)
#define SWA_HAS_ASYNC_LDS 1
#else
#define SWA_HAS_ASYNC_LDS 0
#endif

DEV u16 f2bf(float f) {
  union { float f; unsigned u; } v; v.f = f;
  unsigned r = v.u + 0x7FFFu + ((v.u >> 16) & 1u);   // round-to-nearest-even
  return (u16)(r >> 16);
}

DEV f32x8 zero8() {
  f32x8 z;
#pragma unroll
  for (int i = 0; i < 8; ++i) z[i] = 0.0f;
  return z;
}

DEV f32x8 wmma_bf16(bf16x16 a, bf16x16 b, f32x8 c) {
  // D = A(16x32) * B(32x16) + C, f32 accumulate
  return __builtin_amdgcn_wmma_f32_16x16x32_bf16(
      /*neg_a=*/false, a, /*neg_b=*/false, b, (short)0, c,
      /*reuse_a=*/false, /*reuse_b=*/false);
}

// A-matrix fragment (16x32 bf16): lane l holds row m=l&15;
// element e -> K = e + 16*(l>>4).  Two contiguous 16B chunks.
DEV bf16x16 load_a_frag(const u16* rowp, int k0, int hi) {
  bf16x8 c0 = *(const bf16x8*)(rowp + k0 + 16 * hi);
  bf16x8 c1 = *(const bf16x8*)(rowp + k0 + 16 * hi + 8);
  bf16x16 a;
#pragma unroll
  for (int i = 0; i < 8; ++i) { a[i] = c0[i]; a[i + 8] = c1[i]; }
  return a;
}

DEV void lds_fence() { asm volatile("s_wait_dscnt 0" ::: "memory"); }

// Stage 16 bytes global -> LDS. Async DMA (ASYNCcnt) when available,
// else synchronous load + ds_store fallback.
DEV void stage16(u16* lds_dst, const u16* gsrc) {
#if SWA_HAS_ASYNC_LDS
  __builtin_amdgcn_global_load_async_to_lds_b128(
      (__attribute__((address_space(1))) i32x4v*)(void*)gsrc,
      (__attribute__((address_space(3))) i32x4v*)(void*)lds_dst,
      0, 0);
#else
  *(bf16x8*)lds_dst = *(const bf16x8*)gsrc;
#endif
}

DEV void stage_wait() {
#if SWA_HAS_ASYNC_LDS
  asm volatile("s_wait_asynccnt 0" ::: "memory");
#endif
}

// ---------------------------------------------------------------------------
// f32 -> bf16 (vectorized by 4)
// ---------------------------------------------------------------------------
__global__ void SWA_cvt_bf16(const float* __restrict__ src,
                             u16* __restrict__ dst, int n4) {
  int i = blockIdx.x * blockDim.x + threadIdx.x;
  if (i >= n4) return;
  float4 v = ((const float4*)src)[i];
  uint2 o;
  o.x = (unsigned)f2bf(v.x) | ((unsigned)f2bf(v.y) << 16);
  o.y = (unsigned)f2bf(v.z) | ((unsigned)f2bf(v.w) << 16);
  ((uint2*)dst)[i] = o;
}

// ---------------------------------------------------------------------------
// C(MxN) = A(MxK) * Bw(NxK)^T   (all bf16 in, f32 out)
// Wave computes 16(M) x 64(N); block = 8 waves = 128 rows; K%32==0.
// The shared 64x32 B-tile is DMA-staged into LDS once per block per k-step
// (double buffered, overlapped with WMMA), instead of 8x duplicated loads.
// ---------------------------------------------------------------------------
__global__ __launch_bounds__(256) void SWA_gemm_nt_bf16(
    const u16* __restrict__ A, const u16* __restrict__ Bw,
    float* __restrict__ C, int M, int N, int K) {
  __shared__ __align__(32) u16 btile[2][64 * 32];   // 2 x 4KB

  int tid  = threadIdx.x;
  int lane = tid & 31;
  int wave = tid >> 5;
  int hi = lane >> 4, lo = lane & 15;
  int m0 = blockIdx.x * 128 + wave * 16;
  int n0 = blockIdx.y * 64;

  // staging assignment: 256 threads x 16B = 4KB tile
  int sr = tid >> 2;            // tile row   (0..63) = weight row n0+sr
  int sc = (tid & 3) * 8;       // tile col   (elements, 16B chunks)
  const u16* gsrc = Bw + (size_t)(n0 + sr) * K + sc;
  u16* ldst = &btile[0][0] + sr * 32 + sc;

  f32x8 acc[4];
#pragma unroll
  for (int j = 0; j < 4; ++j) acc[j] = zero8();

  // prologue: stage k-step 0 into buffer 0
  stage16(ldst, gsrc);

  const u16* arow = A + (size_t)(m0 + lo) * K;
  int nk = K >> 5;
  for (int it = 0; it < nk; ++it) {
    int cur = it & 1;
    stage_wait();        // my async stage of btile[cur] has landed
    __syncthreads();     // everyone's has; everyone done reading btile[cur^1]
    if (it + 1 < nk)     // overlap next tile's DMA with this tile's WMMAs
      stage16(&btile[cur ^ 1][0] + sr * 32 + sc, gsrc + ((it + 1) << 5));

    int k0 = it << 5;
    bf16x16 a = load_a_frag(arow, k0, hi);
#pragma unroll
    for (int j = 0; j < 4; ++j) {
      // B fragment: lane l holds column n=n0+16j+(l&15), K=16*(l>>4)+e contig
      bf16x16 b = *(const bf16x16*)&btile[cur][(16 * j + lo) * 32 + 16 * hi];
      acc[j] = wmma_bf16(a, b, acc[j]);
    }
  }

#pragma unroll
  for (int j = 0; j < 4; ++j)
#pragma unroll
    for (int v = 0; v < 8; ++v)
      C[(size_t)(m0 + v + 8 * hi) * N + n0 + 16 * j + lo] = acc[j][v];
}

// ---------------------------------------------------------------------------
// Per-(t,h) RMSNorm (optional) + RoPE + bf16 pack.  One wave per 128-row.
// Optionally writes pre-RoPE normalized f32 (new_kv_k path). d must be 128.
// ---------------------------------------------------------------------------
__global__ __launch_bounds__(256) void SWA_norm_rope(
    const float* __restrict__ X, const float* __restrict__ w,
    u16* __restrict__ outb, float* __restrict__ outf,
    int posBase, int H, int d, int nrows) {
  int row = blockIdx.x * 8 + (threadIdx.x >> 5);
  if (row >= nrows) return;
  int lane = threadIdx.x & 31;
  const float* x = X + (size_t)row * d;
  float4 xv = *(const float4*)(x + lane * 4);
  float y0 = xv.x, y1 = xv.y, y2 = xv.z, y3 = xv.w;

  if (w) {
    float ss = y0 * y0 + y1 * y1 + y2 * y2 + y3 * y3;
#pragma unroll
    for (int m = 16; m >= 1; m >>= 1) ss += __shfl_xor(ss, m, 32);
    float nrm = rsqrtf(ss / (float)d + 1e-6f);
    float4 wv = *(const float4*)(w + lane * 4);
    y0 *= nrm * wv.x; y1 *= nrm * wv.y; y2 *= nrm * wv.z; y3 *= nrm * wv.w;
  }
  if (outf) {
    float4 o; o.x = y0; o.y = y1; o.z = y2; o.w = y3;
    *(float4*)(outf + (size_t)row * d + lane * 4) = o;
  }
  // RoPE: pair i covers dims (2i, 2i+1); this lane holds pairs 2*lane, 2*lane+1
  int t = row / H;
  float pos = (float)(posBase + t);
  const float LOGT = 9.210340371976184f;  // ln(10000)
  int i0 = 2 * lane, i1 = 2 * lane + 1;
  float f0 = __expf(-(2.0f * (float)i0 / (float)d) * LOGT);
  float f1 = __expf(-(2.0f * (float)i1 / (float)d) * LOGT);
  float s0, c0, s1, c1;
  sincosf(pos * f0, &s0, &c0);
  sincosf(pos * f1, &s1, &c1);
  float o0 = y0 * c0 - y1 * s0, o1 = y0 * s0 + y1 * c0;
  float o2 = y2 * c1 - y3 * s1, o3 = y2 * s1 + y3 * c1;
  u16* ob = outb + (size_t)row * d + lane * 4;
  ob[0] = f2bf(o0); ob[1] = f2bf(o1); ob[2] = f2bf(o2); ob[3] = f2bf(o3);
}

// ---------------------------------------------------------------------------
// V path: copy f32 out (new_kv_v) and scatter into d-major transposed bf16
// vT[(h*d+dc)*NKV + jBase + r].
// ---------------------------------------------------------------------------
__global__ __launch_bounds__(256) void SWA_v_pack(
    const float* __restrict__ X, float* __restrict__ outf,
    u16* __restrict__ vT, int jBase, int rows, int D, int NKV) {
  int idx = blockIdx.x * blockDim.x + threadIdx.x;
  if (idx >= rows * D) return;
  int r = idx / D, c = idx % D;
  float x = X[idx];
  if (outf) outf[idx] = x;
  vT[(size_t)c * NKV + jBase + r] = f2bf(x);
}

// ---------------------------------------------------------------------------
// Flash SWA attention: one wave per (head, 16-query tile).
// S = Q K^T via WMMA (bf16), online softmax in f32 via LDS, PV via WMMA.
// ---------------------------------------------------------------------------
__global__ __launch_bounds__(32) void SWA_attn(
    const u16* __restrict__ qb,   // T x D   (rope'd, normalized)
    const u16* __restrict__ kf,   // NKV x D (rope'd; cache rows un-normed)
    const u16* __restrict__ vT,   // (H*d) x NKV  (d-major)
    u16* __restrict__ ob,         // T x D   (bf16 O, pre-Wo)
    const int* __restrict__ chunk_id_p,
    int T, int D, int H, int d, int NKV, int WS) {
  __shared__ float Pt[16][33];
  __shared__ float rowM[16], rowS[16], rowScale[16];

  int lane = threadIdx.x;
  int hi = lane >> 4, lo = lane & 15;
  int t0 = blockIdx.x * 16;
  int h  = blockIdx.y;
  int start_q = chunk_id_p[0] * T;
  int jlowg = WS - start_q; if (jlowg < 0) jlowg = 0;
  float scale = rsqrtf((float)d);

  // Q A-fragments, resident for the whole KV sweep (d=128 -> 4 frags)
  bf16x16 qa[4];
  const u16* qrow = qb + (size_t)(t0 + lo) * D + h * d;
#pragma unroll
  for (int kk = 0; kk < 4; ++kk) qa[kk] = load_a_frag(qrow, kk * 32, hi);

  f32x8 o[8];
#pragma unroll
  for (int c = 0; c < 8; ++c) o[c] = zero8();

  if (lane < 16) { rowM[lane] = -__builtin_inff(); rowS[lane] = 0.0f; }
  lds_fence();

  int jf = t0 + 1; if (jf < jlowg) jf = jlowg;
  int jb = jf & ~31;
  int je = t0 + 15 + WS; if (je > NKV - 1) je = NKV - 1;

  for (int j0 = jb; j0 <= je; j0 += 32) {
    // ---- S tile (16 q x 32 keys) : 8 WMMAs
    f32x8 s0 = zero8(), s1 = zero8();
#pragma unroll
    for (int kk = 0; kk < 4; ++kk) {
      const u16* kbase = kf + (size_t)(j0 + lo) * D + h * d + kk * 32 + 16 * hi;
      bf16x16 b0 = *(const bf16x16*)kbase;
      bf16x16 b1 = *(const bf16x16*)(kbase + (size_t)16 * D);
      s0 = wmma_bf16(qa[kk], b0, s0);
      s1 = wmma_bf16(qa[kk], b1, s1);
    }
#pragma unroll
    for (int v = 0; v < 8; ++v) {
      Pt[v + 8 * hi][lo]      = s0[v];
      Pt[v + 8 * hi][16 + lo] = s1[v];
    }
    lds_fence();

    // ---- online softmax: lane r owns query row r
    if (lane < 16) {
      int r = lane, t = t0 + r;
      int jmin = t + 1; if (jmin < jlowg) jmin = jlowg;
      int jmax = t + WS; if (jmax > NKV - 1) jmax = NKV - 1;
      float mOld = rowM[r], mNew = mOld;
      for (int c = 0; c < 32; ++c) {
        int j = j0 + c;
        if (j >= jmin && j <= jmax) {
          float sv = Pt[r][c] * scale;
          if (sv > mNew) mNew = sv;
        }
      }
      float sc = (mNew == -__builtin_inff()) ? 1.0f : __expf(mOld - mNew);
      float ssum = 0.0f;
      for (int c = 0; c < 32; ++c) {
        int j = j0 + c;
        float e = 0.0f;
        if (j >= jmin && j <= jmax) e = __expf(Pt[r][c] * scale - mNew);
        Pt[r][c] = e;
        ssum += e;
      }
      rowS[r] = rowS[r] * sc + ssum;
      rowM[r] = mNew;
      rowScale[r] = sc;
    }
    lds_fence();

    // ---- rescale O accumulators
#pragma unroll
    for (int v = 0; v < 8; ++v) {
      float sc = rowScale[v + 8 * hi];
#pragma unroll
      for (int c = 0; c < 8; ++c) o[c][v] *= sc;
    }

    // ---- P as A-fragment (16x32), from LDS f32 -> bf16
    bf16x16 pa;
#pragma unroll
    for (int e = 0; e < 16; ++e) {
      int k = e + 16 * hi;
      pa[e] = (__bf16)Pt[lo][k];
    }

    // ---- O += P * V : 8 WMMAs (d = 8 chunks of 16)
#pragma unroll
    for (int c = 0; c < 8; ++c) {
      const u16* vbase = vT + (size_t)(h * d + 16 * c + lo) * NKV + j0 + 16 * hi;
      bf16x16 vb = *(const bf16x16*)vbase;
      o[c] = wmma_bf16(pa, vb, o[c]);
    }
  }

  lds_fence();
#pragma unroll
  for (int v = 0; v < 8; ++v) {
    float inv = 1.0f / rowS[v + 8 * hi];
    int trow = t0 + v + 8 * hi;
#pragma unroll
    for (int c = 0; c < 8; ++c)
      ob[(size_t)trow * D + h * d + 16 * c + lo] = f2bf(o[c][v] * inv);
  }
}

// ---------------------------------------------------------------------------
extern "C" void kernel_launch(void* const* d_in, const int* in_sizes, int n_in,
                              void* d_out, int out_size, void* d_ws,
                              size_t ws_size, hipStream_t stream) {
  const int T = 2048, D = 2048, H = 16, dd = 128, WS = 2048;
  const int NKV = WS + T;

  const float* hs = (const float*)d_in[0];
  const float* kc = (const float*)d_in[1];
  const float* vc = (const float*)d_in[2];
  const float* wq = (const float*)d_in[3];
  const float* wk = (const float*)d_in[4];
  const float* wv = (const float*)d_in[5];
  const float* wo = (const float*)d_in[6];
  const float* qw = (const float*)d_in[7];
  const float* kw = (const float*)d_in[8];
  const int*  cid = (const int*)d_in[9];

  float* out  = (float*)d_out;
  float* newk = out + (size_t)T * D;
  float* newv = newk + (size_t)T * D;

  // ---- workspace carve (~136 MB)
  char* wp = (char*)d_ws;
  auto carve = [&](size_t bytes) -> char* {
    char* p = wp; wp += (bytes + 255) & ~(size_t)255; return p;
  };
  u16*   hb   = (u16*)carve((size_t)T * D * 2);
  u16*   wqb  = (u16*)carve((size_t)D * D * 2);
  u16*   wkb  = (u16*)carve((size_t)D * D * 2);
  u16*   wvb  = (u16*)carve((size_t)D * D * 2);
  u16*   wob  = (u16*)carve((size_t)D * D * 2);
  float* xqf  = (float*)carve((size_t)T * D * 4);
  float* xkf  = (float*)carve((size_t)T * D * 4);
  float* xvf  = (float*)carve((size_t)T * D * 4);
  u16*   qbuf = (u16*)carve((size_t)T * D * 2);
  u16*   kfb  = (u16*)carve((size_t)NKV * D * 2);
  u16*   vTb  = (u16*)carve((size_t)NKV * D * 2);
  u16*   obuf = (u16*)carve((size_t)T * D * 2);

  // 1) f32 -> bf16 conversions
  int n4 = T * D / 4;
  dim3 cb(256), cg((n4 + 255) / 256);
  SWA_cvt_bf16<<<cg, cb, 0, stream>>>(hs, hb, n4);
  SWA_cvt_bf16<<<cg, cb, 0, stream>>>(wq, wqb, n4);
  SWA_cvt_bf16<<<cg, cb, 0, stream>>>(wk, wkb, n4);
  SWA_cvt_bf16<<<cg, cb, 0, stream>>>(wv, wvb, n4);
  SWA_cvt_bf16<<<cg, cb, 0, stream>>>(wo, wob, n4);

  // 2) projections: X @ W^T
  dim3 gb(256), gg(T / 128, D / 64);
  SWA_gemm_nt_bf16<<<gg, gb, 0, stream>>>(hb, wqb, xqf, T, D, D);
  SWA_gemm_nt_bf16<<<gg, gb, 0, stream>>>(hb, wkb, xkf, T, D, D);
  SWA_gemm_nt_bf16<<<gg, gb, 0, stream>>>(hb, wvb, xvf, T, D, D);

  // 3) norm + rope + packing
  int nrows = T * H;
  SWA_norm_rope<<<nrows / 8, 256, 0, stream>>>(xqf, qw, qbuf, nullptr, WS, H, dd, nrows);
  SWA_norm_rope<<<nrows / 8, 256, 0, stream>>>(xkf, kw, kfb + (size_t)WS * D, newk, WS, H, dd, nrows);
  SWA_norm_rope<<<(WS * H) / 8, 256, 0, stream>>>(kc, nullptr, kfb, nullptr, 0, H, dd, WS * H);

  int ne = WS * D;
  SWA_v_pack<<<ne / 256, 256, 0, stream>>>(xvf, newv, vTb, WS, T, D, NKV);
  SWA_v_pack<<<ne / 256, 256, 0, stream>>>(vc, nullptr, vTb, 0, WS, D, NKV);

  // 4) sliding-window flash attention
  SWA_attn<<<dim3(T / 16, H), 32, 0, stream>>>(qbuf, kfb, vTb, obuf, cid,
                                               T, D, H, dd, NKV, WS);

  // 5) output projection
  SWA_gemm_nt_bf16<<<gg, gb, 0, stream>>>(obuf, wob, out, T, D, D);
}
